// GRUBaseline_1675037245540
// MI455X (gfx1250) — matrix operand order repaired and checked by reading, not compiled, not run on hardware
//
#include <hip/hip_runtime.h>
#include <hip/hip_bf16.h>

typedef __attribute__((ext_vector_type(16))) __bf16 v16bf;
typedef __attribute__((ext_vector_type(8)))  __bf16 bf16x8;
typedef __attribute__((ext_vector_type(8)))  float  v8f;
typedef __attribute__((ext_vector_type(4)))  unsigned int u32x4;
typedef __attribute__((ext_vector_type(8)))  int i32x8;
typedef __attribute__((ext_vector_type(4)))  int i32x4;

#define NT    512          // T
#define NB    256          // B
#define DOBS  128
#define DACT  32
#define DHID  512
#define DIN   160          // OBS + ACT
#define D3H   1536         // 3*HID
#define NTB   (NT * NB)
#define LOG2PI_F 1.8378770664093453f
#define LSTR  40           // LDS row stride (bf16): 64B data + 16B pad -> bank-conflict-free
#define TILE_ELEMS ((128 + 64) * LSTR)   // one double-buffer slot: A(128x40) + B(64x40)

// ---------------------------------------------------------------------------
// TDM: issue one 2-D tensor_load_to_lds. Tile is tile_h rows x 32 bf16 cols,
// row stride = stride_elems (bf16), destination LDS rows padded to LSTR via
// D# pad fields (pad_interval code 3 = 64B, pad_amount code 3 = 16B).
// D# packing per CDNA5 ISA (count=1, type=2, dsize=2B).
// This toolchain exposes the 6-arg builtin (extra i32x8 group, zero-filled).
// ---------------------------------------------------------------------------
__device__ __forceinline__ void tdm_load_2d(unsigned lds_byte, const __bf16* gptr,
                                            unsigned width_elems, unsigned rows,
                                            unsigned tile_h, unsigned stride_elems)
{
    unsigned long long ga = (unsigned long long)(size_t)gptr;
    u32x4 g0;
    g0[0] = 1u;                                              // count=1 (valid user D#)
    g0[1] = lds_byte;                                        // lds_addr
    g0[2] = (unsigned)ga;                                    // global_addr[31:0]
    g0[3] = (unsigned)((ga >> 32) & 0x01FFFFFFu)             // global_addr[56:32]
          | 0x80000000u;                                     // type=2 ("image") in bits 127:126
    i32x8 g1;
    g1[0] = 0x06D10000;                                      // data_size=2B | pad_enable | intv=3 | amt=3
    g1[1] = (int)((width_elems & 0xFFFFu) << 16);            // tensor_dim0[15:0]
    g1[2] = (int)((width_elems >> 16) | ((rows & 0xFFFFu) << 16));   // td0[31:16] | td1[15:0]
    g1[3] = (int)((rows >> 16) | (32u << 16));               // td1[31:16] | tile_dim0=32
    g1[4] = (int)(tile_h & 0xFFFFu);                         // tile_dim1 | tile_dim2=0
    g1[5] = (int)stride_elems;                               // tensor_dim0_stride[31:0]
    g1[6] = 0;                                               // stride[47:32]=0 | td1_stride lo = 0
    g1[7] = 0;
    i32x4 z4 = {0, 0, 0, 0};
    i32x8 z8 = {0, 0, 0, 0, 0, 0, 0, 0};
    __builtin_amdgcn_tensor_load_to_lds(g0, g1, z4, z4, z8, 0);
}

// ---------------------------------------------------------------------------
// WMMA GEMM: C[M,N] = act(A[M,K](bf16) @ W[N,K]^T(bf16) + bias(f32))
// MODE 0: f32 out, linear   MODE 1: bf16 out, ELU   MODE 2: bf16 out, linear
// GUARD_N: epilogue store guard for N not multiple of 64 (loads handled by D#).
// 256 threads = 8 wave32; block tile 128x64; wave tile 32x32 (4 WMMAs/K-step).
// Wave 0 drives the TDM; LDS tiles double-buffered, pipelined one K-step ahead.
// Requires M % 128 == 0 and K % 32 == 0 (true for all call sites).
// ---------------------------------------------------------------------------
template <int MODE, bool GUARD_N>
__global__ __launch_bounds__(256)
void wmma_gemm_kernel(const __bf16* __restrict__ A, const __bf16* __restrict__ W,
                      const float* __restrict__ bias, void* __restrict__ Cout,
                      int M, int N, int K, int lda, int ldw)
{
    __shared__ __align__(16) __bf16 sMem[2 * TILE_ELEMS];   // assumed at LDS offset 0

    const int m0   = blockIdx.y * 128;
    const int n0   = blockIdx.x * 64;
    const int tid  = threadIdx.x;
    const int lane = tid & 31;
    const int wv   = tid >> 5;
    const int wm   = (wv >> 1) * 32;     // 4 waves down M
    const int wn   = (wv & 1)  * 32;     // 2 waves across N
    const int l16  = lane & 15;
    const int kh   = lane >> 4;
    const bool issuer = (wv == 0);

    const unsigned bRows = GUARD_N ? (unsigned)(N - n0 < 64 ? N - n0 : 64) : 64u;

    const v8f vz = {0.f,0.f,0.f,0.f,0.f,0.f,0.f,0.f};
    v8f acc[2][2] = {{vz, vz}, {vz, vz}};

    const int nk = K >> 5;   // K/32 tiles

    // prologue: tile 0 -> buffer 0
    if (issuer) {
        tdm_load_2d(0u,                    A + (size_t)m0 * lda, (unsigned)K, 128u, 128u, (unsigned)lda);
        tdm_load_2d(128u * LSTR * 2u,      W + (size_t)n0 * ldw, (unsigned)K, bRows, bRows, (unsigned)ldw);
    }

    for (int kt = 0; kt < nk; ++kt) {
        const int buf = kt & 1;
        if (issuer) {
            if (kt + 1 < nk) {
                const int k1 = (kt + 1) << 5;
                const unsigned base = (unsigned)((buf ^ 1) * TILE_ELEMS * 2);
                tdm_load_2d(base,                 A + (size_t)m0 * lda + k1,
                            (unsigned)(K - k1), 128u, 128u, (unsigned)lda);
                tdm_load_2d(base + 128u * LSTR * 2u, W + (size_t)n0 * ldw + k1,
                            (unsigned)(K - k1), bRows, bRows, (unsigned)ldw);
                __builtin_amdgcn_s_wait_tensorcnt(2);   // retire tile kt (in-order completion)
            } else {
                __builtin_amdgcn_s_wait_tensorcnt(0);
            }
        }
        __syncthreads();    // tile kt visible to all waves

        const __bf16* sA = &sMem[buf * TILE_ELEMS];
        const __bf16* sB = sA + 128 * LSTR;

        // ---- fragments per gfx1250 wave32 WMMA layouts ----
        // A 16x32: lane<16 holds M=l16, K = {kh*8..+7, 16+kh*8..+7}
        v16bf afr[2], bfr[2];
#pragma unroll
        for (int i = 0; i < 2; ++i) {
            const __bf16* p = &sA[(wm + 16 * i + l16) * LSTR + kh * 8];
            bf16x8 lo = *(const bf16x8*)p;
            bf16x8 hi = *(const bf16x8*)(p + 16);
            afr[i] = __builtin_shufflevector(lo, hi, 0,1,2,3,4,5,6,7,8,9,10,11,12,13,14,15);
        }
        // B 32x16: lane holds N=l16, K = {kh*8..+7, 16+kh*8..+7} (K-contiguous in sB row)
#pragma unroll
        for (int j = 0; j < 2; ++j) {
            const __bf16* p = &sB[(wn + 16 * j + l16) * LSTR + kh * 8];
            bf16x8 lo = *(const bf16x8*)p;
            bf16x8 hi = *(const bf16x8*)(p + 16);
            bfr[j] = __builtin_shufflevector(lo, hi, 0,1,2,3,4,5,6,7,8,9,10,11,12,13,14,15);
        }
#pragma unroll
        for (int i = 0; i < 2; ++i)
#pragma unroll
            for (int j = 0; j < 2; ++j)
                acc[i][j] = __builtin_amdgcn_wmma_f32_16x16x32_bf16(
                    false, afr[i], false, bfr[j], (short)0, acc[i][j], false, false);

        __syncthreads();    // all waves done reading buf before TDM overwrites it
    }

    // ---- epilogue: C layout lane l16=N, M = r + 8*kh ----
#pragma unroll
    for (int i = 0; i < 2; ++i) {
#pragma unroll
        for (int j = 0; j < 2; ++j) {
#pragma unroll
            for (int r = 0; r < 8; ++r) {
                const int gm = m0 + wm + 16 * i + kh * 8 + r;
                const int gn = n0 + wn + 16 * j + l16;
                if (!GUARD_N || gn < N) {
                    float v = acc[i][j][r] + bias[gn];
                    if (MODE == 1) v = (v > 0.f) ? v : (__expf(v) - 1.f);   // ELU
                    if (MODE == 0) ((float*)Cout)[(size_t)gm * N + gn] = v;
                    else           ((__bf16*)Cout)[(size_t)gm * N + gn] = (__bf16)v;
                }
            }
        }
    }
}

// ---------------------------------------------------------------------------
// GRU gating: r=sig(gi_r+gh_r), z=sig(gi_z+gh_z), n=tanh(gi_n + r*gh_n),
// h' = (1-z)*n + z*h.  Emits new h (f32 + bf16) and the hs[] slot.
// ---------------------------------------------------------------------------
__global__ __launch_bounds__(256)
void gru_gate_kernel(const __bf16* __restrict__ gi, const float* __restrict__ gh,
                     float* __restrict__ h, __bf16* __restrict__ hb,
                     __bf16* __restrict__ hs_slot)
{
    int i = blockIdx.x * blockDim.x + threadIdx.x;
    if (i >= NB * DHID) return;
    int b = i / DHID, j = i % DHID;
    const __bf16* gib = gi + (size_t)b * D3H;
    const float*  ghb = gh + (size_t)b * D3H;
    float r  = 1.f / (1.f + __expf(-((float)gib[j]            + ghb[j])));
    float z  = 1.f / (1.f + __expf(-((float)gib[j + DHID]     + ghb[j + DHID])));
    float n  = tanhf((float)gib[j + 2 * DHID] + r * ghb[j + 2 * DHID]);
    float hn = (1.f - z) * n + z * h[i];
    h[i] = hn;
    __bf16 hv = (__bf16)hn;
    hb[i] = hv;
    hs_slot[i] = hv;
}

// ---------------------------------------------------------------------------
__global__ void f32_to_bf16_kernel(const float* __restrict__ in, __bf16* __restrict__ o, int n)
{
    int i = blockIdx.x * blockDim.x + threadIdx.x;
    if (i < n) o[i] = (__bf16)in[i];
}

__global__ void build_x_kernel(const float* __restrict__ obs, const float* __restrict__ act,
                               __bf16* __restrict__ X)
{
    int i = blockIdx.x * blockDim.x + threadIdx.x;
    if (i >= NTB * DIN) return;
    int row = i / DIN, c = i % DIN;
    float v = (c < DOBS) ? obs[(size_t)row * DOBS + c]
                         : act[(size_t)row * DACT + (c - DOBS)];
    X[i] = (__bf16)v;
}

__global__ void build_x0_kernel(const float* __restrict__ obs, __bf16* __restrict__ X0)
{
    int i = blockIdx.x * blockDim.x + threadIdx.x;
    if (i >= NB * DIN) return;
    int b = i / DIN, c = i % DIN;
    float v = (c < DOBS) ? obs[(size_t)b * DOBS + c] : 0.f;   // x0 = [obs[0], zeros]
    X0[i] = (__bf16)v;
}

__global__ void zero_init_kernel(float* __restrict__ h, __bf16* __restrict__ hb,
                                 float* __restrict__ loss)
{
    int i = blockIdx.x * blockDim.x + threadIdx.x;
    if (i < NB * DHID) { h[i] = 0.f; hb[i] = (__bf16)0.f; }
    if (i == 0) loss[0] = 0.f;
}

// ---------------------------------------------------------------------------
// Loss: sum of -logprob for (obs vs pre_obs), (action vs pre_action),
// (action vs pre_reward broadcast -- faithful to the reference bug), / (T*B)
// ---------------------------------------------------------------------------
__global__ __launch_bounds__(256)
void loss_kernel(const float* __restrict__ obs, const float* __restrict__ act,
                 const float* __restrict__ po, const float* __restrict__ pa,
                 const float* __restrict__ pr, float* __restrict__ loss)
{
    __shared__ float red[256];
    int row = blockIdx.x * 256 + threadIdx.x;   // < NTB
    float s = 0.f;
    const float* o = obs + (size_t)row * DOBS;
    const float* p = po  + (size_t)row * DOBS;
    for (int c = 0; c < DOBS; ++c) { float d = o[c] - p[c]; s += 0.5f * d * d + 0.5f * LOG2PI_F; }
    const float* a = act + (size_t)row * DACT;
    const float* q = pa  + (size_t)row * DACT;
    for (int c = 0; c < DACT; ++c) { float d = a[c] - q[c]; s += 0.5f * d * d + 0.5f * LOG2PI_F; }
    float mu = pr[row];
    for (int c = 0; c < DACT; ++c) { float d = a[c] - mu;   s += 0.5f * d * d + 0.5f * LOG2PI_F; }
    red[threadIdx.x] = s;
    __syncthreads();
    for (int st = 128; st > 0; st >>= 1) {
        if (threadIdx.x < st) red[threadIdx.x] += red[threadIdx.x + st];
        __syncthreads();
    }
    if (threadIdx.x == 0) atomicAdd(loss, red[0] * (1.f / (float)NTB));
}

// ---------------------------------------------------------------------------
extern "C" void kernel_launch(void* const* d_in, const int* in_sizes, int n_in,
                              void* d_out, int out_size, void* d_ws, size_t ws_size,
                              hipStream_t stream)
{
    (void)in_sizes; (void)n_in; (void)out_size; (void)ws_size;
    const float* obs    = (const float*)d_in[0];
    const float* action = (const float*)d_in[1];
    // d_in[2] = reward: unused by the reference loss
    const float* w_ih = (const float*)d_in[3];
    const float* w_hh = (const float*)d_in[4];
    const float* b_ih = (const float*)d_in[5];
    const float* b_hh = (const float*)d_in[6];
    const float* ow0 = (const float*)d_in[7];  const float* ob0 = (const float*)d_in[8];
    const float* ow1 = (const float*)d_in[9];  const float* ob1 = (const float*)d_in[10];
    const float* ow2 = (const float*)d_in[11]; const float* ob2 = (const float*)d_in[12];
    const float* aw0 = (const float*)d_in[13]; const float* ab0 = (const float*)d_in[14];
    const float* aw1 = (const float*)d_in[15]; const float* ab1 = (const float*)d_in[16];
    const float* aw2 = (const float*)d_in[17]; const float* ab2 = (const float*)d_in[18];
    const float* rw0 = (const float*)d_in[19]; const float* rb0 = (const float*)d_in[20];
    const float* rw1 = (const float*)d_in[21]; const float* rb1 = (const float*)d_in[22];
    const float* rw2 = (const float*)d_in[23]; const float* rb2 = (const float*)d_in[24];

    float* out     = (float*)d_out;
    float* pre_obs = out + 1;
    float* pre_act = pre_obs + (size_t)NTB * DOBS;
    float* pre_rew = pre_act + (size_t)NTB * DACT;

    // bump allocator over workspace (all offsets 256B aligned)
    char* ws = (char*)d_ws;
    size_t off = 0;
    auto alloc = [&](size_t bytes) -> void* {
        void* p = ws + off;
        off = (off + bytes + 255) & ~(size_t)255;
        return p;
    };

    __bf16* Xbf  = (__bf16*)alloc((size_t)NTB * DIN * 2);       // [T*B,160]
    __bf16* X0bf = (__bf16*)alloc((size_t)NB * DIN * 2);
    __bf16* wihb = (__bf16*)alloc((size_t)D3H * DIN * 2);
    __bf16* whhb = (__bf16*)alloc((size_t)D3H * DHID * 2);
    __bf16* ow0b = (__bf16*)alloc((size_t)DHID * DHID * 2);
    __bf16* ow1b = (__bf16*)alloc((size_t)DHID * DHID * 2);
    __bf16* ow2b = (__bf16*)alloc((size_t)DOBS * DHID * 2);
    __bf16* aw0b = (__bf16*)alloc((size_t)DHID * DHID * 2);
    __bf16* aw1b = (__bf16*)alloc((size_t)DHID * DHID * 2);
    __bf16* aw2b = (__bf16*)alloc((size_t)DACT * DHID * 2);
    __bf16* rw0b = (__bf16*)alloc((size_t)DHID * DHID * 2);
    __bf16* rw1b = (__bf16*)alloc((size_t)DHID * DHID * 2);
    __bf16* rw2b = (__bf16*)alloc((size_t)1 * DHID * 2);
    __bf16* GIb  = (__bf16*)alloc((size_t)NTB * D3H * 2);       // hoisted input gates, bf16
    __bf16* gi0b = (__bf16*)alloc((size_t)NB * D3H * 2);        // bootstrap-step gi
    float*  GH   = (float*) alloc((size_t)NB * D3H * 4);        // per-step h-side gates
    float*  Hf   = (float*) alloc((size_t)NB * DHID * 4);       // h state f32
    __bf16* Hb   = (__bf16*)alloc((size_t)NB * DHID * 2);       // h state bf16 (GEMM A)
    __bf16* HSb  = (__bf16*)alloc((size_t)NTB * DHID * 2);      // emitted hs sequence
    __bf16* Z1   = (__bf16*)alloc((size_t)NTB * DHID * 2);      // MLP hidden 1
    __bf16* Z2   = (__bf16*)alloc((size_t)NTB * DHID * 2);      // MLP hidden 2

    dim3 blk(256);
    // h = 0, loss = 0
    zero_init_kernel<<<(NB * DHID + 255) / 256, blk, 0, stream>>>(Hf, Hb, out);

    // weight conversions f32 -> bf16
    auto conv = [&](const float* s, __bf16* d, size_t n) {
        f32_to_bf16_kernel<<<(unsigned)((n + 255) / 256), blk, 0, stream>>>(s, d, (int)n);
    };
    conv(w_ih, wihb, (size_t)D3H * DIN);
    conv(w_hh, whhb, (size_t)D3H * DHID);
    conv(ow0, ow0b, (size_t)DHID * DHID); conv(ow1, ow1b, (size_t)DHID * DHID);
    conv(ow2, ow2b, (size_t)DOBS * DHID);
    conv(aw0, aw0b, (size_t)DHID * DHID); conv(aw1, aw1b, (size_t)DHID * DHID);
    conv(aw2, aw2b, (size_t)DACT * DHID);
    conv(rw0, rw0b, (size_t)DHID * DHID); conv(rw1, rw1b, (size_t)DHID * DHID);
    conv(rw2, rw2b, (size_t)DHID);

    build_x_kernel<<<(NTB * DIN + 255) / 256, blk, 0, stream>>>(obs, action, Xbf);
    build_x0_kernel<<<(NB * DIN + 255) / 256, blk, 0, stream>>>(obs, X0bf);

    // Hoisted input-side gates for ALL steps: GI = X @ w_ih^T + b_ih   (K=160=5*32)
    wmma_gemm_kernel<2, false><<<dim3(D3H / 64, NTB / 128), blk, 0, stream>>>(
        Xbf, wihb, b_ih, GIb, NTB, D3H, DIN, DIN, DIN);
    // bootstrap gi for x0
    wmma_gemm_kernel<2, false><<<dim3(D3H / 64, NB / 128), blk, 0, stream>>>(
        X0bf, wihb, b_ih, gi0b, NB, D3H, DIN, DIN, DIN);

    // h0 = GRU(x0, 0):  gh from zeroed h, then gate; emit hs[0] = h0
    wmma_gemm_kernel<0, false><<<dim3(D3H / 64, NB / 128), blk, 0, stream>>>(
        Hb, whhb, b_hh, GH, NB, D3H, DHID, DHID, DHID);
    gru_gate_kernel<<<(NB * DHID + 255) / 256, blk, 0, stream>>>(gi0b, GH, Hf, Hb, HSb);

    // Sequential scan: hs[t+1] = GRU(x_t, hs[t]) for t = 0..T-2 (h_T never emitted)
    for (int t = 0; t < NT - 1; ++t) {
        wmma_gemm_kernel<0, false><<<dim3(D3H / 64, NB / 128), blk, 0, stream>>>(
            Hb, whhb, b_hh, GH, NB, D3H, DHID, DHID, DHID);
        gru_gate_kernel<<<(NB * DHID + 255) / 256, blk, 0, stream>>>(
            GIb + (size_t)t * NB * D3H, GH, Hf, Hb, HSb + (size_t)(t + 1) * NB * DHID);
    }

    // Three decoder MLPs over all T*B rows (ELU hidden layers fused in epilogue)
    dim3 gBig(DHID / 64, NTB / 128);
    // obs head -> pre_obs
    wmma_gemm_kernel<1, false><<<gBig, blk, 0, stream>>>(HSb, ow0b, ob0, Z1, NTB, DHID, DHID, DHID, DHID);
    wmma_gemm_kernel<1, false><<<gBig, blk, 0, stream>>>(Z1, ow1b, ob1, Z2, NTB, DHID, DHID, DHID, DHID);
    wmma_gemm_kernel<0, false><<<dim3(DOBS / 64, NTB / 128), blk, 0, stream>>>(
        Z2, ow2b, ob2, pre_obs, NTB, DOBS, DHID, DHID, DHID);
    // action head -> pre_action
    wmma_gemm_kernel<1, false><<<gBig, blk, 0, stream>>>(HSb, aw0b, ab0, Z1, NTB, DHID, DHID, DHID, DHID);
    wmma_gemm_kernel<1, false><<<gBig, blk, 0, stream>>>(Z1, aw1b, ab1, Z2, NTB, DHID, DHID, DHID, DHID);
    wmma_gemm_kernel<0, true><<<dim3(1, NTB / 128), blk, 0, stream>>>(
        Z2, aw2b, ab2, pre_act, NTB, DACT, DHID, DHID, DHID);
    // reward head -> pre_reward (N=1)
    wmma_gemm_kernel<1, false><<<gBig, blk, 0, stream>>>(HSb, rw0b, rb0, Z1, NTB, DHID, DHID, DHID, DHID);
    wmma_gemm_kernel<1, false><<<gBig, blk, 0, stream>>>(Z1, rw1b, rb1, Z2, NTB, DHID, DHID, DHID, DHID);
    wmma_gemm_kernel<0, true><<<dim3(1, NTB / 128), blk, 0, stream>>>(
        Z2, rw2b, rb2, pre_rew, NTB, 1, DHID, DHID, DHID);

    // Final loss reduction into out[0]
    loss_kernel<<<NTB / 256, blk, 0, stream>>>(obs, action, pre_obs, pre_act, pre_rew, out);
}
